// ImplicitPhysicsAttention_37744172597584
// MI455X (gfx1250) — matrix operand order repaired and checked by reading, not compiled, not run on hardware
//
#include <hip/hip_runtime.h>

typedef __attribute__((ext_vector_type(16))) __bf16 v16bf;
typedef __attribute__((ext_vector_type(8)))  __bf16 v8bf;
typedef __attribute__((ext_vector_type(8)))  float  v8f;
typedef __attribute__((ext_vector_type(4)))  unsigned int u32x4;
typedef __attribute__((ext_vector_type(8)))  unsigned int u32x8;

static __device__ inline v16bf join16(v8bf lo, v8bf hi) {
    union { v16bf v; v8bf h[2]; } u;
    u.h[0] = lo; u.h[1] = hi;
    return u.v;
}
static __device__ inline v8bf ld8(const __bf16* p) {
    return *reinterpret_cast<const v8bf*>(p);
}
static __device__ inline void st8(__bf16* p, v8bf v) {
    *reinterpret_cast<v8bf*>(p) = v;
}
static __device__ inline v8f wmma_bf16(v16bf a, v16bf b, v8f c) {
    // D = A(16x32 bf16) * B(32x16 bf16) + C(16x16 f32)
    return __builtin_amdgcn_wmma_f32_16x16x32_bf16(
        false, a, false, b, (short)0, c, false, false);
}

// Async 16-byte copy global -> LDS (VGLOBAL GLOBAL_LOAD_ASYNC_TO_LDS_B128,
// tracked with ASYNCcnt). LDS destination address = low 32 bits of the
// generic pointer (generic = aperture | lds_offset on AMDGPU).
static __device__ inline void async16(const __bf16* gsrc, __bf16* ldst) {
    unsigned laddr = (unsigned)(size_t)ldst;
    asm volatile("global_load_async_to_lds_b128 %0, %1, off"
                 :: "v"(laddr), "v"(gsrc)
                 : "memory");
}
static __device__ inline void wait_async0() {
    asm volatile("s_wait_asynccnt 0x0" ::: "memory");
}

#define NB   8
#define NN   1024
#define ND   512
#define NH   8
#define NHD  64
#define NLAT 32
#define NENVK 80

// ---------------------------------------------------------------------------
// f32 -> bf16 pack (flat)
// ---------------------------------------------------------------------------
__global__ void k_pack(const float* __restrict__ in, __bf16* __restrict__ out, int n) {
    int i = blockIdx.x * blockDim.x + threadIdx.x;
    if (i < n) out[i] = (__bf16)in[i];
}

// f32 (R x C, row-major) -> bf16 transposed (C x R, row-major)
__global__ void k_packT(const float* __restrict__ in, __bf16* __restrict__ out, int R, int C) {
    int i = blockIdx.x * blockDim.x + threadIdx.x;
    if (i < R * C) {
        int r = i / C, c = i % C;
        out[(size_t)c * R + r] = (__bf16)in[i];
    }
}

// ---------------------------------------------------------------------------
// Conv branch: spatial mean of SAME 3x3 conv collapses to 9 region sums.
// Produces per-(b,h) score scale: 0.125 * sca / (att + 1e-6)
// ---------------------------------------------------------------------------
__global__ __launch_bounds__(256) void k_conv_phy(
    const float* __restrict__ keys, const float* __restrict__ conv_w,
    const float* __restrict__ conv_b, const float* __restrict__ phy_w,
    const float* __restrict__ phy_b, float* __restrict__ sscale)
{
    int b = blockIdx.x;
    int t = threadIdx.x;
    __shared__ float accL[NLAT];
    __shared__ float mv[NLAT];
    __shared__ float phy[2 * NH];
    if (t < NLAT) accL[t] = 0.f;
    __syncthreads();

    float loc[NLAT];
#pragma unroll
    for (int l = 0; l < NLAT; l++) loc[l] = 0.f;

    for (int c = t; c < ND; c += 256) {
        const float* Kb = keys + (size_t)b * NN * ND + c;
        float T = 0.f, R0 = 0.f, R31 = 0.f, C0 = 0.f, C31 = 0.f;
        for (int p = 0; p < NN; p++) {
            float v = Kb[(size_t)p * ND];
            T += v;
            int y = p >> 5, x = p & 31;
            if (y == 0)  R0  += v;
            if (y == 31) R31 += v;
            if (x == 0)  C0  += v;
            if (x == 31) C31 += v;
        }
        float k00    = Kb[0];
        float k0_31  = Kb[(size_t)31 * ND];
        float k31_0  = Kb[(size_t)992 * ND];
        float k31_31 = Kb[(size_t)1023 * ND];
        for (int dy = 0; dy < 3; dy++) {
            for (int dx = 0; dx < 3; dx++) {
                float S = T;
                if (dy == 0) S -= R31;
                if (dy == 2) S -= R0;
                if (dx == 0) S -= C31;
                if (dx == 2) S -= C0;
                if (dy == 0 && dx == 0) S += k31_31;
                if (dy == 0 && dx == 2) S += k31_0;
                if (dy == 2 && dx == 0) S += k0_31;
                if (dy == 2 && dx == 2) S += k00;
                const float* wt = conv_w + ((size_t)(dy * 3 + dx) * ND + c) * NLAT;
#pragma unroll
                for (int l = 0; l < NLAT; l++) loc[l] += wt[l] * S;
            }
        }
    }
#pragma unroll
    for (int l = 0; l < NLAT; l++) atomicAdd(&accL[l], loc[l]);
    __syncthreads();
    if (t < NLAT) mv[t] = conv_b[t] + accL[t] * (1.f / 1024.f);
    __syncthreads();
    if (t < 2 * NH) {
        float a = phy_b[t];
        for (int l = 0; l < NLAT; l++) a += mv[l] * phy_w[l * (2 * NH) + t];
        phy[t] = a;
    }
    __syncthreads();
    if (t < NH) sscale[b * NH + t] = 0.125f * phy[NH + t] / (phy[t] + 1e-6f);
}

// ---------------------------------------------------------------------------
// Tiled bf16 GEMM: C = A(MxK) * W(KxN) + bias, W supplied transposed (N x K).
// LDS tiles filled with GLOBAL_LOAD_ASYNC_TO_LDS_B128 (ASYNCcnt).
// Optional f32 C and/or bf16 C2 outputs. 128x128x32 tiles, 8 waves (2x4).
// ---------------------------------------------------------------------------
__global__ __launch_bounds__(256) void k_gemm(
    const __bf16* __restrict__ A, int lda,
    const __bf16* __restrict__ WT, int ldw,
    const float* __restrict__ bias,
    float* __restrict__ C, int ldc,
    __bf16* __restrict__ C2, int ldc2,
    int M, int K)
{
    __shared__ __bf16 sA[128 * 32];
    __shared__ __bf16 sB[128 * 32];

    int t = threadIdx.x;
    int m0 = blockIdx.y * 128, n0 = blockIdx.x * 128;
    int wv = t >> 5, lane = t & 31, half = lane >> 4, l15 = lane & 15;
    int wm = (wv & 1) * 64, wn = (wv >> 1) * 32;
    int kba = half * 8, kbb = half * 16;

    v8f acc[4][2];
#pragma unroll
    for (int mt = 0; mt < 4; mt++)
#pragma unroll
        for (int nt = 0; nt < 2; nt++)
#pragma unroll
            for (int i = 0; i < 8; i++) acc[mt][nt][i] = 0.f;

    int r = t >> 1, cs = (t & 1) * 16;
    int grow = m0 + r;
    if (grow >= M) grow = M - 1;   // clamp; clamped rows never stored
    int nk = K >> 5;
    for (int kt = 0; kt < nk; kt++) {
        int k0 = kt * 32;
        __syncthreads();
        // async-stage A and W^T tiles straight into LDS
        {
            const __bf16* ga = A + (size_t)grow * lda + k0 + cs;
            async16(ga,     &sA[r * 32 + cs]);
            async16(ga + 8, &sA[r * 32 + cs + 8]);
            const __bf16* gb = WT + (size_t)(n0 + r) * ldw + k0 + cs;
            async16(gb,     &sB[r * 32 + cs]);
            async16(gb + 8, &sB[r * 32 + cs + 8]);
        }
        if (kt + 1 < nk) {
            __builtin_prefetch(A  + (size_t)grow * lda + k0 + 32 + cs, 0, 1);
            __builtin_prefetch(WT + (size_t)(n0 + r) * ldw + k0 + 32 + cs, 0, 1);
        }
        wait_async0();
        __syncthreads();

        v16bf af[4];
#pragma unroll
        for (int mt = 0; mt < 4; mt++) {
            const __bf16* pa = &sA[(wm + mt * 16 + l15) * 32 + kba];
            af[mt] = join16(ld8(pa), ld8(pa + 16));
        }
#pragma unroll
        for (int nt = 0; nt < 2; nt++) {
            const __bf16* pb = &sB[(wn + nt * 16 + l15) * 32 + kba];
            v16bf bf = join16(ld8(pb), ld8(pb + 16));
#pragma unroll
            for (int mt = 0; mt < 4; mt++)
                acc[mt][nt] = wmma_bf16(af[mt], bf, acc[mt][nt]);
        }
    }

#pragma unroll
    for (int mt = 0; mt < 4; mt++) {
#pragma unroll
        for (int nt = 0; nt < 2; nt++) {
#pragma unroll
            for (int i = 0; i < 8; i++) {
                int row = m0 + wm + mt * 16 + i + half * 8;
                int col = n0 + wn + nt * 16 + l15;
                if (row < M) {
                    float v = acc[mt][nt][i] + bias[col];
                    if (C)  C[(size_t)row * ldc + col] = v;
                    if (C2) C2[(size_t)row * ldc2 + col] = (__bf16)v;
                }
            }
        }
    }
}

// ---------------------------------------------------------------------------
// Flash attention per (b,h): 64-row Q tiles, 4 waves x 16 rows, online softmax.
// K tile staged via TDM tensor_load_to_lds (TENSORcnt), issued by wave 0.
// ---------------------------------------------------------------------------
__global__ __launch_bounds__(128) void k_flash_attn(
    const __bf16* __restrict__ qb, const __bf16* __restrict__ kb,
    const __bf16* __restrict__ vb, const float* __restrict__ sscale,
    __bf16* __restrict__ att_out)
{
    int bh = blockIdx.y;
    int b = bh >> 3, h = bh & 7;
    int qt = blockIdx.x;
    int t = threadIdx.x;
    int w = t >> 5, lane = t & 31, half = lane >> 4, l15 = lane & 15;
    int kba = half * 8, kbb = half * 16;

    __shared__ __bf16 Kt[64 * 64];       // [key][dim]
    __shared__ __bf16 Vt[64 * 64];       // [dim][key] (transposed)
    __shared__ __bf16 Pw[4 * 16 * 64];   // per-wave P scratch

    float ss = sscale[bh];

    // preload Q fragments (stay resident across kv loop)
    int qrow = qt * 64 + w * 16 + l15;
    size_t qbase = ((size_t)b * NN + qrow) * ND + h * NHD;
    v16bf qf[2];
#pragma unroll
    for (int kk = 0; kk < 2; kk++)
        qf[kk] = join16(ld8(qb + qbase + kk * 32 + kba),
                        ld8(qb + qbase + kk * 32 + 16 + kba));

    float mr[8], lr[8];
    v8f oacc[4];
#pragma unroll
    for (int i = 0; i < 8; i++) { mr[i] = -1e30f; lr[i] = 0.f; }
#pragma unroll
    for (int jn = 0; jn < 4; jn++)
#pragma unroll
        for (int i = 0; i < 8; i++) oacc[jn][i] = 0.f;

    unsigned kt_lds = (unsigned)(size_t)(&Kt[0]);

    for (int kt = 0; kt < 16; kt++) {
        int k0 = kt * 64;
        __syncthreads();
        // ---- K tile via Tensor Data Mover (one wave issues the DMA) ----
        if (w == 0) {
            unsigned long long ga =
                (unsigned long long)(size_t)(kb + ((size_t)b * NN + k0) * ND + h * NHD);
            // D# group 0: count=1, lds_addr, 57b global byte addr, type=2
            u32x4 g0;
            g0[0] = (unsigned)__builtin_amdgcn_readfirstlane((int)1u);
            g0[1] = (unsigned)__builtin_amdgcn_readfirstlane((int)kt_lds);
            g0[2] = (unsigned)__builtin_amdgcn_readfirstlane((int)(unsigned)(ga & 0xffffffffu));
            g0[3] = (unsigned)__builtin_amdgcn_readfirstlane(
                        (int)(((unsigned)(ga >> 32) & 0x01ffffffu) | (2u << 30)));
            // D# group 1: data_size=2B; tensor_dim0=512, tensor_dim1=4096;
            // tile 64x64; tensor_dim0_stride=512 elements
            u32x8 g1;
            g1[0] = (unsigned)__builtin_amdgcn_readfirstlane((int)(1u << 16));
            g1[1] = (unsigned)__builtin_amdgcn_readfirstlane((int)(512u << 16));
            g1[2] = (unsigned)__builtin_amdgcn_readfirstlane((int)(4096u << 16));
            g1[3] = (unsigned)__builtin_amdgcn_readfirstlane((int)(64u << 16));
            g1[4] = (unsigned)__builtin_amdgcn_readfirstlane((int)64u);
            g1[5] = (unsigned)__builtin_amdgcn_readfirstlane((int)512u);
            g1[6] = (unsigned)__builtin_amdgcn_readfirstlane((int)0u);
            g1[7] = (unsigned)__builtin_amdgcn_readfirstlane((int)0u);
            asm volatile("tensor_load_to_lds %0, %1" :: "s"(g0), "s"(g1) : "memory");
            asm volatile("s_wait_tensorcnt 0x0" ::: "memory");
        }
        // ---- V tile, transposed (TDM cannot transpose) ----
        {
            int dim = t >> 1, kh = (t & 1) * 32;
            for (int i = 0; i < 32; i++)
                Vt[dim * 64 + kh + i] =
                    vb[((size_t)b * NN + k0 + kh + i) * ND + h * NHD + dim];
        }
        __syncthreads();

        // S = Q @ K^T (4 key groups of 16)
        v8f s[4];
#pragma unroll
        for (int kg = 0; kg < 4; kg++) {
            v8f a;
#pragma unroll
            for (int i = 0; i < 8; i++) a[i] = 0.f;
#pragma unroll
            for (int kk = 0; kk < 2; kk++) {
                const __bf16* p = &Kt[(kg * 16 + l15) * 64 + kk * 32 + kba];
                a = wmma_bf16(qf[kk], join16(ld8(p), ld8(p + 16)), a);
            }
            s[kg] = a;
        }
#pragma unroll
        for (int kg = 0; kg < 4; kg++)
#pragma unroll
            for (int i = 0; i < 8; i++) s[kg][i] *= ss;

        // online softmax (row = i + 8*half; reduce across 16-lane halves)
        float alpha[8];
#pragma unroll
        for (int i = 0; i < 8; i++) {
            float mx = fmaxf(fmaxf(s[0][i], s[1][i]), fmaxf(s[2][i], s[3][i]));
            for (int m = 1; m < 16; m <<= 1) mx = fmaxf(mx, __shfl_xor(mx, m, 32));
            float nm = fmaxf(mr[i], mx);
            alpha[i] = __expf(mr[i] - nm);
            mr[i] = nm;
        }
#pragma unroll
        for (int i = 0; i < 8; i++) {
            float rs = 0.f;
#pragma unroll
            for (int kg = 0; kg < 4; kg++) {
                float e = __expf(s[kg][i] - mr[i]);
                s[kg][i] = e; rs += e;
            }
            for (int m = 1; m < 16; m <<= 1) rs += __shfl_xor(rs, m, 32);
            lr[i] = lr[i] * alpha[i] + rs;
#pragma unroll
            for (int jn = 0; jn < 4; jn++) oacc[jn][i] *= alpha[i];
        }

        // P -> wave-private LDS (bf16) to re-shape into A-fragment layout
        __bf16* pw = &Pw[w * 16 * 64];
#pragma unroll
        for (int kg = 0; kg < 4; kg++)
#pragma unroll
            for (int i = 0; i < 8; i++)
                pw[(i + half * 8) * 64 + kg * 16 + l15] = (__bf16)s[kg][i];
        asm volatile("s_wait_dscnt 0x0" ::: "memory");

        // O += P @ V
#pragma unroll
        for (int kk = 0; kk < 2; kk++) {
            const __bf16* pa = &pw[l15 * 64 + kk * 32 + kba];
            v16bf pf = join16(ld8(pa), ld8(pa + 16));
#pragma unroll
            for (int jn = 0; jn < 4; jn++) {
                const __bf16* pv = &Vt[(jn * 16 + l15) * 64 + kk * 32 + kba];
                oacc[jn] = wmma_bf16(pf, join16(ld8(pv), ld8(pv + 16)), oacc[jn]);
            }
        }
    }

#pragma unroll
    for (int jn = 0; jn < 4; jn++) {
#pragma unroll
        for (int i = 0; i < 8; i++) {
            int n = qt * 64 + w * 16 + i + half * 8;
            int dim = h * NHD + jn * 16 + l15;
            att_out[((size_t)b * NN + n) * ND + dim] = (__bf16)(oacc[jn][i] / lr[i]);
        }
    }
}

// ---------------------------------------------------------------------------
// Env attention: 8192 groups of (8 queries x 88 keys x 64 dims), f32 VALU.
// ---------------------------------------------------------------------------
__global__ __launch_bounds__(256) void k_env_attn(
    const __bf16* __restrict__ qe, const __bf16* __restrict__ ke,
    const __bf16* __restrict__ ve, const float* __restrict__ envp,
    const float* __restrict__ lam_p, __bf16* __restrict__ env_out)
{
    int g = blockIdx.x;                 // g = n*8 + h
    int n = g >> 3, hk = g & 7, hv = g >> 10;
    int t = threadIdx.x;

    __shared__ float qs[8 * 64], ks[8 * 64], vs[8 * 64];
    __shared__ float ek[NENVK * 64], ev[NENVK * 64];
    __shared__ float sc[8 * 88];

    float lam = *lam_p;
    for (int i = t; i < 512; i += 256) {
        int bb = i >> 6, d = i & 63;
        size_t base = ((size_t)bb * NN + n) * ND + hk * NHD + d;
        qs[i] = (float)qe[base];
        ks[i] = (float)ke[base];
        vs[i] = (float)ve[base];
    }
    for (int i = t; i < NENVK * 64; i += 256) {
        int e = i >> 6, d = i & 63;
        ek[i] = envp[(size_t)e * ND + hk * NHD + d];
        ev[i] = envp[(size_t)e * ND + hv * NHD + d];
    }
    __syncthreads();

    const float scale = 0.125f;
    for (int idx = t; idx < 8 * 88; idx += 256) {
        int bb = idx / 88, j = idx % 88;
        float acc = 0.f;
        if (j < 8) {
            for (int d = 0; d < 64; d++) acc += qs[bb * 64 + d] * ks[j * 64 + d];
            acc *= scale;
        } else {
            int e = j - 8;
            for (int d = 0; d < 64; d++) acc += qs[bb * 64 + d] * ek[e * 64 + d];
            acc *= lam * scale;
        }
        sc[idx] = acc;
    }
    __syncthreads();

    int wv = t >> 5, lane = t & 31;
    if (wv < 8) {
        float mx = -1e30f;
        for (int j = lane; j < 88; j += 32) mx = fmaxf(mx, sc[wv * 88 + j]);
        for (int m = 1; m < 32; m <<= 1) mx = fmaxf(mx, __shfl_xor(mx, m, 32));
        float sm = 0.f;
        for (int j = lane; j < 88; j += 32) {
            float e = __expf(sc[wv * 88 + j] - mx);
            sc[wv * 88 + j] = e; sm += e;
        }
        for (int m = 1; m < 32; m <<= 1) sm += __shfl_xor(sm, m, 32);
        float inv = 1.f / sm;
        for (int j = lane; j < 88; j += 32) sc[wv * 88 + j] *= inv;
    }
    __syncthreads();

    for (int i = t; i < 512; i += 256) {
        int bb = i >> 6, d = i & 63;
        float acc = 0.f;
        for (int j = 0; j < 8; j++) acc += sc[bb * 88 + j] * vs[j * 64 + d];
        for (int e = 0; e < NENVK; e++) acc += sc[bb * 88 + 8 + e] * ev[e * 64 + d];
        env_out[((size_t)bb * NN + n) * ND + hk * NHD + d] = (__bf16)acc;
    }
}

// ---------------------------------------------------------------------------
// Final gated blend: y = sigmoid(gpre) * out + (1 - sigmoid(gpre)) * oenv
// ---------------------------------------------------------------------------
__global__ void k_final(const float* __restrict__ outf, const float* __restrict__ oenvf,
                        const float* __restrict__ gpre, float* __restrict__ y, int n)
{
    int i = blockIdx.x * blockDim.x + threadIdx.x;
    if (i < n) {
        float g = 1.f / (1.f + __expf(-gpre[i]));
        y[i] = g * outf[i] + (1.f - g) * oenvf[i];
    }
}

// ---------------------------------------------------------------------------
extern "C" void kernel_launch(void* const* d_in, const int* in_sizes, int n_in,
                              void* d_out, int out_size, void* d_ws, size_t ws_size,
                              hipStream_t stream) {
    const float* queries  = (const float*)d_in[0];
    const float* keys     = (const float*)d_in[1];
    const float* values   = (const float*)d_in[2];
    const float* conv_w   = (const float*)d_in[3];
    const float* conv_b   = (const float*)d_in[4];
    const float* phy_w    = (const float*)d_in[5];
    const float* phy_b    = (const float*)d_in[6];
    const float* to_out_w = (const float*)d_in[7];
    const float* to_out_b = (const float*)d_in[8];
    const float* env_keys = (const float*)d_in[9];
    const float* env_w    = (const float*)d_in[10];
    const float* env_b    = (const float*)d_in[11];
    const float* lam      = (const float*)d_in[12];
    const float* q_w      = (const float*)d_in[13];
    const float* q_b      = (const float*)d_in[14];
    const float* k_w      = (const float*)d_in[15];
    const float* k_b      = (const float*)d_in[16];
    const float* v_w      = (const float*)d_in[17];
    const float* v_b      = (const float*)d_in[18];
    const float* out_w    = (const float*)d_in[19];
    const float* out_b    = (const float*)d_in[20];
    const float* gate_w   = (const float*)d_in[21];
    const float* gate_b   = (const float*)d_in[22];

    const size_t MQ = (size_t)NB * NN;        // 8192 rows
    const size_t DSZ = ND;

    char* base = (char*)d_ws;
    size_t off = 0;
    auto alloc = [&](size_t bytes) -> char* {
        char* p = base + off;
        off = (off + bytes + 255) & ~(size_t)255;
        return p;
    };

    __bf16* qb     = (__bf16*)alloc(MQ * DSZ * 2);
    __bf16* kb     = (__bf16*)alloc(MQ * DSZ * 2);
    __bf16* vb     = (__bf16*)alloc(MQ * DSZ * 2);
    __bf16* envkb  = (__bf16*)alloc((size_t)NENVK * DSZ * 2);
    __bf16* WqT    = (__bf16*)alloc(DSZ * DSZ * 2);
    __bf16* WkT    = (__bf16*)alloc(DSZ * DSZ * 2);
    __bf16* WvT    = (__bf16*)alloc(DSZ * DSZ * 2);
    __bf16* WtoT   = (__bf16*)alloc(DSZ * DSZ * 2);
    __bf16* WoutT  = (__bf16*)alloc(DSZ * DSZ * 2);
    __bf16* WenvT  = (__bf16*)alloc(DSZ * DSZ * 2);
    __bf16* WgT    = (__bf16*)alloc(DSZ * 2 * DSZ * 2);   // 512 x 1024
    __bf16* attO   = (__bf16*)alloc(MQ * DSZ * 2);
    __bf16* qe     = (__bf16*)alloc(MQ * DSZ * 2);
    __bf16* ke     = (__bf16*)alloc(MQ * DSZ * 2);
    __bf16* ve     = (__bf16*)alloc(MQ * DSZ * 2);
    __bf16* envO   = (__bf16*)alloc(MQ * DSZ * 2);
    __bf16* gateA  = (__bf16*)alloc(MQ * 2 * DSZ * 2);    // [out | oenv] bf16
    float*  sscale = (float*)alloc(NB * NH * 4);
    float*  envp   = (float*)alloc((size_t)NENVK * DSZ * 4);
    float*  outF   = (float*)alloc(MQ * DSZ * 4);
    float*  oenvF  = (float*)alloc(MQ * DSZ * 4);
    float*  gpre   = (float*)alloc(MQ * DSZ * 4);

    const int TPB = 256;
    int nBig = (int)(MQ * DSZ);
    dim3 gBig((nBig + TPB - 1) / TPB);

    // 1) conv/phy scalars
    k_conv_phy<<<NB, 256, 0, stream>>>(keys, conv_w, conv_b, phy_w, phy_b, sscale);

    // 2) pack activations + weights (weights transposed)
    k_pack<<<gBig, TPB, 0, stream>>>(queries, qb, nBig);
    k_pack<<<gBig, TPB, 0, stream>>>(keys,    kb, nBig);
    k_pack<<<gBig, TPB, 0, stream>>>(values,  vb, nBig);
    k_pack<<<(NENVK * ND + TPB - 1) / TPB, TPB, 0, stream>>>(env_keys, envkb, NENVK * ND);
    dim3 gW((ND * ND + TPB - 1) / TPB);
    k_packT<<<gW, TPB, 0, stream>>>(q_w,      WqT,  ND, ND);
    k_packT<<<gW, TPB, 0, stream>>>(k_w,      WkT,  ND, ND);
    k_packT<<<gW, TPB, 0, stream>>>(v_w,      WvT,  ND, ND);
    k_packT<<<gW, TPB, 0, stream>>>(to_out_w, WtoT, ND, ND);
    k_packT<<<gW, TPB, 0, stream>>>(out_w,    WoutT, ND, ND);
    k_packT<<<gW, TPB, 0, stream>>>(env_w,    WenvT, ND, ND);
    k_packT<<<(2 * ND * ND + TPB - 1) / TPB, TPB, 0, stream>>>(gate_w, WgT, 2 * ND, ND);

    // 3) main attention (flash, WMMA + TDM)
    k_flash_attn<<<dim3(NN / 64, NB * NH), 128, 0, stream>>>(qb, kb, vb, sscale, attO);

    // 4) projections (WMMA GEMM, async LDS staging)
    dim3 gGemm(ND / 128, (unsigned)(MQ / 128));
    k_gemm<<<gGemm, 256, 0, stream>>>(qb, ND, WqT, ND, q_b, nullptr, 0, qe, ND, (int)MQ, ND);
    k_gemm<<<gGemm, 256, 0, stream>>>(kb, ND, WkT, ND, k_b, nullptr, 0, ke, ND, (int)MQ, ND);
    k_gemm<<<gGemm, 256, 0, stream>>>(vb, ND, WvT, ND, v_b, nullptr, 0, ve, ND, (int)MQ, ND);
    k_gemm<<<dim3(ND / 128, 1), 256, 0, stream>>>(envkb, ND, WenvT, ND, env_b,
                                                  envp, ND, nullptr, 0, NENVK, ND);

    // 5) env attention (reads envp f32)
    k_env_attn<<<(unsigned)MQ, 256, 0, stream>>>(qe, ke, ve, envp, lam, envO);

    // 6) output projections; bf16 copies land in gateA = [out | oenv]
    k_gemm<<<gGemm, 256, 0, stream>>>(attO, ND, WtoT, ND, to_out_b,
                                      outF, ND, gateA, 2 * ND, (int)MQ, ND);
    k_gemm<<<gGemm, 256, 0, stream>>>(envO, ND, WoutT, ND, out_b,
                                      oenvF, ND, gateA + ND, 2 * ND, (int)MQ, ND);

    // 7) gate GEMM over concatenated K=1024
    k_gemm<<<gGemm, 256, 0, stream>>>(gateA, 2 * ND, WgT, 2 * ND, gate_b,
                                      gpre, ND, nullptr, 0, (int)MQ, 2 * ND);

    // 8) final blend
    k_final<<<gBig, TPB, 0, stream>>>(outF, oenvF, gpre, (float*)d_out, nBig);
}
